// LSTM_net_33569464385918
// MI455X (gfx1250) — matrix operand order repaired and checked by reading, not compiled, not run on hardware
//
#include <hip/hip_runtime.h>
#include <hip/hip_bf16.h>
#include <math.h>

// Problem dims
#define B_   512
#define C_   22
#define T_   1000
#define H_   64
#define G_   256   // 4*H
#define CP_  32    // C padded to 32 for one f16 WMMA K-chunk
#define NCLS 4
#define GSTR 260   // padded gate row stride (floats) to avoid LDS bank conflicts

typedef __attribute__((ext_vector_type(16))) _Float16 v16h;
typedef __attribute__((ext_vector_type(8)))  _Float16 v8h;
typedef __attribute__((ext_vector_type(8)))  float    v8f;

// Workspace layout (bytes):
//   [0)                 xs_f16  : B*T*32 f16 = 32,768,000
//   [32768000)          whh_f16 : 256*64 f16 =     32,768
//   [32800768)          wih_f16 : 256*32 f16 =     16,384
//   [32817152)          bias    : 256 f32    =      1,024
#define WS_XS   0
#define WS_WHH  32768000
#define WS_WIH  32800768
#define WS_BIAS 32817152

// ---------------------------------------------------------------------------
// A-fragment pack for 16-bit WMMA A (16x32), per ISA table:
// lanes 0-15 hold K=0..7 (VGPR0-3) and K=16..23 (VGPR4-7)
// lanes16-31 hold K=8..15          and K=24..31
__device__ __forceinline__ v16h combine8(v8h lo, v8h hi) {
  union { v16h v; v8h h[2]; } u;
  u.h[0] = lo;
  u.h[1] = hi;
  return u.v;
}
__device__ __forceinline__ v16h pack_a(const _Float16* base, int lhalf) {
  return combine8(*(const v8h*)(base + lhalf * 8),
                  *(const v8h*)(base + 16 + lhalf * 8));
}

__device__ __forceinline__ float sigm(float x) {
  return 1.0f / (1.0f + __expf(-x));
}
__device__ __forceinline__ float tanh_fast(float x) {
  x = fminf(15.0f, fmaxf(-15.0f, x));
  float e = __expf(2.0f * x);
  return (e - 1.0f) / (e + 1.0f);
}

// ---------------------------------------------------------------------------
// Kernel 0: weight prep. whh/wih -> f16 (wih zero-padded C 22->32), bias sum.
__global__ void lstm_prep_kernel(const float* __restrict__ W_ih,
                                 const float* __restrict__ W_hh,
                                 const float* __restrict__ b_ih,
                                 const float* __restrict__ b_hh,
                                 _Float16* __restrict__ whh,
                                 _Float16* __restrict__ wih,
                                 float* __restrict__ bias) {
  int i = blockIdx.x * 256 + threadIdx.x;
  if (i < G_ * H_) whh[i] = (_Float16)W_hh[i];
  int j = i - G_ * H_;
  if (j >= 0 && j < G_ * CP_) {
    int g = j >> 5, c = j & 31;
    wih[j] = (c < C_) ? (_Float16)W_ih[g * C_ + c] : (_Float16)0.0f;
  }
  int k = i - (G_ * H_ + G_ * CP_);
  if (k >= 0 && k < G_) bias[k] = b_ih[k] + b_hh[k];
}

// ---------------------------------------------------------------------------
// Kernel 1: transpose x [B][C][T] f32 -> xs [B][T][32] f16 (zero-padded C).
__global__ void lstm_xpose_kernel(const float* __restrict__ x,
                                  _Float16* __restrict__ xs) {
  int b = blockIdx.x;
  int t = blockIdx.y * 256 + threadIdx.x;
  if (t >= T_) return;
  v8h part[4];
#pragma unroll
  for (int i = 0; i < 4; ++i) {
#pragma unroll
    for (int k2 = 0; k2 < 8; ++k2) {
      int c = i * 8 + k2;
      part[i][k2] = (c < C_) ? (_Float16)x[(b * C_ + c) * T_ + t]
                             : (_Float16)0.0f;
    }
  }
  _Float16* dst = xs + ((size_t)b * T_ + t) * CP_;
#pragma unroll
  for (int i = 0; i < 4; ++i) *(v8h*)(dst + i * 8) = part[i];
}

// ---------------------------------------------------------------------------
// Kernel 2: fused LSTM recurrence + FC accumulation + softmax.
// Grid: 32 blocks (batch tiles of 16) x 512 threads (16 waves).
// Per wave: ONE N-tile of the 256 gate columns (3 dependent WMMAs/step);
// WMMA order is h0 -> h1 -> x so both h ds_load pairs must be live together
// (parallel issue, single dscnt wait) and the pipelined x global load waits
// only at the last WMMA.
extern "C" __global__ void __launch_bounds__(512, 1)
lstm_core_kernel(const _Float16* __restrict__ xs,
                 const _Float16* __restrict__ whh,
                 const _Float16* __restrict__ wih,
                 const float* __restrict__ bias,
                 const float* __restrict__ W_fc,
                 const float* __restrict__ b_fc,
                 float* __restrict__ out) {
  __shared__ alignas(32) _Float16 h_lds[16 * H_];       // 2 KB
  __shared__ alignas(16) float    gates[16 * GSTR];     // ~16.6 KB
  __shared__ float                logits[16 * NCLS];    // 256 B

  const int tid   = threadIdx.x;
  const int b0    = blockIdx.x * 16;
  const int lane  = tid & 31;
  const int wave  = tid >> 5;        // 0..15
  const int lhalf = lane >> 4;       // 0 or 1
  const int l15   = lane & 15;

  // zero h state and logits
  for (int i = tid; i < 16 * H_; i += 512) h_lds[i] = (_Float16)0.0f;
  if (tid < 16 * NCLS) logits[tid] = 0.0f;

  // B-fragments for this wave's N-tile (persistent in VGPRs)
  const int n0 = wave * 16;
  const v16h bi  = pack_a(wih + (n0 + l15) * CP_, lhalf);
  const v16h bh0 = pack_a(whh + (n0 + l15) * H_ + 0, lhalf);
  const v16h bh1 = pack_a(whh + (n0 + l15) * H_ + 32, lhalf);
  const float bias_n = bias[n0 + l15];

  // Elementwise-phase thread ownership: j fixed per thread, 2 rows m.
  const int j     = tid & 63;
  const int mbase = tid >> 6;        // 0..7 ; m = mbase + 8*p
  float cst[2]  = {0.f, 0.f};
  float facc[2][NCLS] = {};

  const _Float16* xrow = xs + (size_t)(b0 + l15) * T_ * CP_;
  const _Float16* hrow = h_lds + l15 * H_;
  const float*    wfcj = W_fc + j;

  __syncthreads();

  // software pipeline: x fragment for step t preloaded during step t-1
  v16h ax = pack_a(xrow, lhalf);

  for (int t = 0; t < T_; ++t) {
    // ---- phase A: gate GEMM via WMMA ----------------------------------
    // Both h fragments are consumed by the first two WMMAs, so all four
    // ds_load_b128 must be issued up front into distinct registers.
    const v16h ah0 = combine8(*(const v8h*)(hrow + 0  + lhalf * 8),
                              *(const v8h*)(hrow + 16 + lhalf * 8));
    const v16h ah1 = combine8(*(const v8h*)(hrow + 32 + lhalf * 8),
                              *(const v8h*)(hrow + 48 + lhalf * 8));

    v8f acc = {bias_n, bias_n, bias_n, bias_n, bias_n, bias_n, bias_n, bias_n};
    acc = __builtin_amdgcn_wmma_f32_16x16x32_f16(false, ah0, false, bh0,
                                                 (short)0, acc, false, false);
    acc = __builtin_amdgcn_wmma_f32_16x16x32_f16(false, ah1, false, bh1,
                                                 (short)0, acc, false, false);
    acc = __builtin_amdgcn_wmma_f32_16x16x32_f16(false, ax,  false, bi,
                                                 (short)0, acc, false, false);

    // C/D layout: VGPR r, lanes 0-15 -> (M=r, N=l15); lanes 16-31 -> (M=r+8)
    const int mrow = lhalf * 8;
#pragma unroll
    for (int r = 0; r < 8; ++r)
      gates[(mrow + r) * GSTR + n0 + l15] = acc[r];
    __syncthreads();

    // ---- phase B: elementwise LSTM cell + fused FC --------------------
    // Issue next step's x load here: it has all of phase B plus the
    // trailing barrier plus next phase A's h-WMMAs before first use.
    // (Last-iter overread lands in the weight region of the same
    // workspace allocation; value never consumed.)
    __builtin_prefetch(xrow + (t + 8) * CP_, 0, 1);   // global_prefetch_b8
    v16h ax_next = pack_a(xrow + (t + 1) * CP_, lhalf);

    const float* wfc = wfcj + t * H_;
#pragma unroll
    for (int p = 0; p < 2; ++p) {
      const int m = mbase + p * 8;
      const float gi = gates[m * GSTR + 0   + j];
      const float gf = gates[m * GSTR + 64  + j];
      const float gg = gates[m * GSTR + 128 + j];
      const float go = gates[m * GSTR + 192 + j];
      const float iv = sigm(gi);
      const float fv = sigm(gf);
      const float gv = tanh_fast(gg);
      const float ov = sigm(go);
      const float c  = fv * cst[p] + iv * gv;
      cst[p] = c;
      const float h = ov * tanh_fast(c);
      h_lds[m * H_ + j] = (_Float16)h;
#pragma unroll
      for (int cl = 0; cl < NCLS; ++cl)
        facc[p][cl] += h * wfc[(size_t)cl * (H_ * T_)];
    }
    ax = ax_next;
    __syncthreads();
  }

  // ---- reduce FC partials over j, then softmax ------------------------
#pragma unroll
  for (int p = 0; p < 2; ++p) {
    const int m = mbase + p * 8;
#pragma unroll
    for (int cl = 0; cl < NCLS; ++cl)
      atomicAdd(&logits[m * NCLS + cl], facc[p][cl]);
  }
  __syncthreads();

  if (tid < 16) {
    float l0 = logits[tid * NCLS + 0] + b_fc[0];
    float l1 = logits[tid * NCLS + 1] + b_fc[1];
    float l2 = logits[tid * NCLS + 2] + b_fc[2];
    float l3 = logits[tid * NCLS + 3] + b_fc[3];
    float mx = fmaxf(fmaxf(l0, l1), fmaxf(l2, l3));
    float e0 = __expf(l0 - mx), e1 = __expf(l1 - mx);
    float e2 = __expf(l2 - mx), e3 = __expf(l3 - mx);
    float inv = 1.0f / (e0 + e1 + e2 + e3);
    float* o = out + (size_t)(b0 + tid) * NCLS;
    o[0] = e0 * inv; o[1] = e1 * inv; o[2] = e2 * inv; o[3] = e3 * inv;
  }
}

// ---------------------------------------------------------------------------
extern "C" void kernel_launch(void* const* d_in, const int* in_sizes, int n_in,
                              void* d_out, int out_size, void* d_ws, size_t ws_size,
                              hipStream_t stream) {
  const float* x    = (const float*)d_in[0];
  const float* W_ih = (const float*)d_in[1];
  const float* W_hh = (const float*)d_in[2];
  const float* b_ih = (const float*)d_in[3];
  const float* b_hh = (const float*)d_in[4];
  const float* W_fc = (const float*)d_in[5];
  const float* b_fc = (const float*)d_in[6];
  float* out = (float*)d_out;

  char* ws = (char*)d_ws;
  _Float16* xs   = (_Float16*)(ws + WS_XS);
  _Float16* whh  = (_Float16*)(ws + WS_WHH);
  _Float16* wih  = (_Float16*)(ws + WS_WIH);
  float*    bias = (float*)   (ws + WS_BIAS);

  // prep: 16384 + 8192 + 256 = 24832 work items
  lstm_prep_kernel<<<97, 256, 0, stream>>>(W_ih, W_hh, b_ih, b_hh, whh, wih, bias);
  // transpose x -> [B][T][32] f16
  lstm_xpose_kernel<<<dim3(B_, 4), 256, 0, stream>>>(x, xs);
  // fused recurrence + FC + softmax
  lstm_core_kernel<<<B_ / 16, 512, 0, stream>>>(xs, whh, wih, bias, W_fc, b_fc, out);
}
